// E_STFGNN_17497696764294
// MI455X (gfx1250) — matrix operand order repaired and checked by reading, not compiled, not run on hardware
//
#include <hip/hip_runtime.h>

typedef __attribute__((ext_vector_type(16))) _Float16 v16h;
typedef __attribute__((ext_vector_type(8)))  _Float16 v8h;
typedef __attribute__((ext_vector_type(4)))  _Float16 v4h;
typedef __attribute__((ext_vector_type(8)))  float    v8f;

#define DM    64
#define TT    12
#define NNODES 20000
#define NROWS (NNODES * TT)     // 240000 (divisible by 16)
#define NNZ_E 320000

// ---------------------------------------------------------------------------
// WMMA: D = A(16x32 f16) * B(32x16 f16) + C(f32), wave32
// ---------------------------------------------------------------------------
__device__ __forceinline__ v8f wmma32(v16h a, v16h b, v8f c) {
  return __builtin_amdgcn_wmma_f32_16x16x32_f16(false, a, false, b,
                                                (short)0, c, false, false);
}

// A fragment from row-major f16 LDS buffer: lane l holds
// M = l&15, K = ((h>=8)?16:0) + 8*(l>>4) + (h&7)  ->  two contiguous 8-half
// runs per lane -> 2x ds_load_b128.
__device__ __forceinline__ v16h load_a_lds(const _Float16* src, int ld, int k0) {
  int lane = threadIdx.x & 31;
  int m    = lane & 15;
  int ksel = (lane >> 4) * 8;
  const _Float16* p = src + m * ld + k0 + ksel;
  v8h lo = *(const v8h*)p;
  v8h hi = *(const v8h*)(p + 16);
  v16h a;
#pragma unroll
  for (int h = 0; h < 8; ++h) { a[h] = lo[h]; a[8 + h] = hi[h]; }
  return a;
}

// B fragment from fragment-packed ("swizzled") LDS: tile (kc,nt) stores
// lane-major 512 halfs; one aligned 32B load per lane -> 2x ds_load_b128.
__device__ __forceinline__ v16h load_b_sw(const _Float16* base, int ntn, int kc, int nt) {
  int lane = threadIdx.x & 31;
  return *(const v16h*)(base + (((kc * ntn + nt) << 9) + (lane << 4)));
}

// A fragment directly from row-major global f32 (ld, k0 multiples of 8):
// two 32B runs (4x global_load_b128) + cvt.
__device__ __forceinline__ v16h load_a_g32(const float* rowbase, int ld, int k0) {
  int lane = threadIdx.x & 31;
  int m    = lane & 15;
  int ksel = (lane >> 4) * 8;
  const float* p = rowbase + (long)m * ld + k0 + ksel;
  float4 x0 = ((const float4*)p)[0];
  float4 x1 = ((const float4*)p)[1];
  float4 y0 = ((const float4*)(p + 16))[0];
  float4 y1 = ((const float4*)(p + 16))[1];
  v16h a;
  a[0] = (_Float16)x0.x; a[1] = (_Float16)x0.y; a[2] = (_Float16)x0.z; a[3] = (_Float16)x0.w;
  a[4] = (_Float16)x1.x; a[5] = (_Float16)x1.y; a[6] = (_Float16)x1.z; a[7] = (_Float16)x1.w;
  a[8] = (_Float16)y0.x; a[9] = (_Float16)y0.y; a[10] = (_Float16)y0.z; a[11] = (_Float16)y0.w;
  a[12] = (_Float16)y1.x; a[13] = (_Float16)y1.y; a[14] = (_Float16)y1.z; a[15] = (_Float16)y1.w;
  return a;
}

// A fragment from global f32 where only K < kmax (<=16) is valid; K>=16 zero.
__device__ __forceinline__ v16h load_a_g32_k(const float* rowbase, int ld, int kmax) {
  int lane = threadIdx.x & 31;
  int m    = lane & 15;
  int ksel = (lane >> 4) * 8;
  v16h a;
#pragma unroll
  for (int h = 0; h < 16; ++h) a[h] = (_Float16)0.f;
  if (ksel < kmax) {
    const float* p = rowbase + (long)m * ld + ksel;
    float4 x0 = ((const float4*)p)[0];
    float4 x1 = ((const float4*)p)[1];
    a[0] = (_Float16)x0.x; a[1] = (_Float16)x0.y; a[2] = (_Float16)x0.z; a[3] = (_Float16)x0.w;
    a[4] = (_Float16)x1.x; a[5] = (_Float16)x1.y; a[6] = (_Float16)x1.z; a[7] = (_Float16)x1.w;
  }
  return a;
}

__device__ __forceinline__ v8f addbias(v8f d, const float* bias, int n0) {
  int n = threadIdx.x & 15;
  float b = bias[n0 + n];
#pragma unroll
  for (int i = 0; i < 8; ++i) d[i] += b;
  return d;
}

// Store D fragment (16x16 f32) into row-major f16 LDS buffer; optional ReLU.
__device__ __forceinline__ void store_d16(_Float16* dst, int ld, int n0, v8f d, bool relu) {
  int lane = threadIdx.x & 31;
  int n  = lane & 15;
  int mh = (lane >> 4) * 8;
#pragma unroll
  for (int i = 0; i < 8; ++i) {
    float v = d[i];
    if (relu) v = v > 0.f ? v : 0.f;
    dst[(mh + i) * ld + n0 + n] = (_Float16)v;
  }
}

// Fill a fragment-packed B buffer from row-major Kv x N f32 weights,
// zero-padding K up to Kpad (multiple of 32).
__device__ __forceinline__ void fill_b_sw(_Float16* dst, const float* w,
                                          int Kpad, int N, int Kv, int nthreads) {
  int total = Kpad * N;
  int ntn = N >> 4;
  for (int i = threadIdx.x; i < total; i += nthreads) {
    int tile = i >> 9, rem = i & 511;
    int lane = rem >> 4, h = rem & 15;
    int kc = tile / ntn, nt = tile % ntn;
    int K = kc * 32 + ((h >> 3) << 4) + ((lane >> 4) << 3) + (h & 7);
    int n = nt * 16 + (lane & 15);
    dst[i] = (K < Kv) ? (_Float16)w[K * N + n] : (_Float16)0.f;
  }
}

// ---------------------------------------------------------------------------
// Kernel 1: fused edge-MLP + weather-MLP + concat + comb -> H (f32)
// 128 threads = 4 waves; one 16-row tile of (N*T) per wave.
// ---------------------------------------------------------------------------
__global__ void k_embed(const float* __restrict__ Xe, const float* __restrict__ Xw,
                        const float* __restrict__ ew1, const float* __restrict__ eb1,
                        const float* __restrict__ ew2, const float* __restrict__ eb2,
                        const float* __restrict__ ww1, const float* __restrict__ wb1,
                        const float* __restrict__ ww2, const float* __restrict__ wb2,
                        const float* __restrict__ cw,  const float* __restrict__ cb,
                        float* __restrict__ H) {
  __shared__ __attribute__((aligned(32))) _Float16 sE1[32 * 32];    // Kpad=32,N=32
  __shared__ __attribute__((aligned(32))) _Float16 sW1[32 * 32];
  __shared__ __attribute__((aligned(32))) _Float16 sE2[32 * 64];
  __shared__ __attribute__((aligned(32))) _Float16 sW2[32 * 64];
  __shared__ __attribute__((aligned(32))) _Float16 sCW[128 * 64];
  __shared__ float sbE1[32], sbE2[64], sbW1[32], sbW2[64], sbC[64];
  __shared__ __attribute__((aligned(32))) _Float16 stHid[4][16 * 32];
  __shared__ __attribute__((aligned(32))) _Float16 stCat[4][16 * 128];

  int tid = threadIdx.x;
  fill_b_sw(sE1, ew1, 32, 32, 16, 128);
  fill_b_sw(sW1, ww1, 32, 32, 8, 128);
  fill_b_sw(sE2, ew2, 32, 64, 32, 128);
  fill_b_sw(sW2, ww2, 32, 64, 32, 128);
  fill_b_sw(sCW, cw, 128, 64, 128, 128);
  for (int i = tid; i < 32; i += 128) { sbE1[i] = eb1[i]; sbW1[i] = wb1[i]; }
  for (int i = tid; i < 64; i += 128) { sbE2[i] = eb2[i]; sbW2[i] = wb2[i]; sbC[i] = cb[i]; }
  __syncthreads();

  int wave = tid >> 5, lane = tid & 31;
  long tile = (long)blockIdx.x * 4 + wave;
  if (tile * 16 >= NROWS) return;
  long row0 = tile * 16;

  _Float16* Hid = stHid[wave];
  _Float16* Cat = stCat[wave];

  // --- edge path: (16x16) @ (16x32) relu @ (32x64) -> Cat[:,0:64] ---
  v16h aE = load_a_g32_k(Xe + row0 * 16, 16, 16);
#pragma unroll
  for (int nt = 0; nt < 2; ++nt) {
    v8f acc = {};
    acc = wmma32(aE, load_b_sw(sE1, 2, 0, nt), acc);
    acc = addbias(acc, sbE1, nt * 16);
    store_d16(Hid, 32, nt * 16, acc, true);
  }
  v16h aH0 = load_a_lds(Hid, 32, 0);
#pragma unroll
  for (int nt = 0; nt < 4; ++nt) {
    v8f acc = {};
    acc = wmma32(aH0, load_b_sw(sE2, 4, 0, nt), acc);
    acc = addbias(acc, sbE2, nt * 16);
    store_d16(Cat, 128, nt * 16, acc, false);
  }

  // --- weather path: (16x8) @ (8x32) relu @ (32x64) -> Cat[:,64:128] ---
  v16h aW = load_a_g32_k(Xw + row0 * 8, 8, 8);
#pragma unroll
  for (int nt = 0; nt < 2; ++nt) {
    v8f acc = {};
    acc = wmma32(aW, load_b_sw(sW1, 2, 0, nt), acc);
    acc = addbias(acc, sbW1, nt * 16);
    store_d16(Hid, 32, nt * 16, acc, true);
  }
  v16h aH1 = load_a_lds(Hid, 32, 0);
#pragma unroll
  for (int nt = 0; nt < 4; ++nt) {
    v8f acc = {};
    acc = wmma32(aH1, load_b_sw(sW2, 4, 0, nt), acc);
    acc = addbias(acc, sbW2, nt * 16);
    store_d16(Cat, 128, 64 + nt * 16, acc, false);
  }

  // --- comb: (16x128) @ (128x64), relu -> H f32 ---
  int n = lane & 15, mh = (lane >> 4) * 8;
#pragma unroll
  for (int nt = 0; nt < 4; ++nt) {
    v8f acc = {};
#pragma unroll
    for (int kc = 0; kc < 4; ++kc) {
      v16h a = load_a_lds(Cat, 128, kc * 32);
      acc = wmma32(a, load_b_sw(sCW, 4, kc, nt), acc);
    }
    acc = addbias(acc, sbC, nt * 16);
#pragma unroll
    for (int i = 0; i < 8; ++i) {
      float v = acc[i];
      v = v > 0.f ? v : 0.f;
      H[(row0 + mh + i) * DM + nt * 16 + n] = v;
    }
  }
}

// ---------------------------------------------------------------------------
// Kernel 2: HW = H @ W (64x64), M = 240000. Grid-stride per wave + prefetch.
// ---------------------------------------------------------------------------
__global__ void k_gemm64(const float* __restrict__ Hin, const float* __restrict__ W,
                         float* __restrict__ Hout) {
  __shared__ __attribute__((aligned(32))) _Float16 sW[64 * 64];   // kc=2, nt=4
  int tid = threadIdx.x;
  fill_b_sw(sW, W, 64, 64, 64, 128);
  __syncthreads();

  int wave = tid >> 5, lane = tid & 31;
  int n = lane & 15, mh = (lane >> 4) * 8;
  long wstride = (long)gridDim.x * 4;
  for (long tile = (long)blockIdx.x * 4 + wave; tile * 16 < NROWS; tile += wstride) {
    const float* rb = Hin + tile * 16 * DM;
    long nxt = tile + wstride;
    if (nxt * 16 < NROWS)
      __builtin_prefetch(Hin + nxt * 16 * DM, 0, 1);   // global_prefetch_b8
    v16h a0 = load_a_g32(rb, 64, 0);
    v16h a1 = load_a_g32(rb, 64, 32);
#pragma unroll
    for (int nt = 0; nt < 4; ++nt) {
      v8f acc = {};
      acc = wmma32(a0, load_b_sw(sW, 4, 0, nt), acc);
      acc = wmma32(a1, load_b_sw(sW, 4, 1, nt), acc);
#pragma unroll
      for (int i = 0; i < 8; ++i)
        Hout[(tile * 16 + mh + i) * DM + nt * 16 + n] = acc[i];
    }
  }
}

// ---------------------------------------------------------------------------
// Kernel 3: sparse scatter: Hgc[r] += v * HW[c], per (edge, t).
// Hgc (61 MB) is L2-resident (192 MB L2) so atomics resolve in L2.
// ---------------------------------------------------------------------------
__global__ void k_scatter(const int* __restrict__ rows, const int* __restrict__ cols,
                          const float* __restrict__ vals,
                          const float* __restrict__ HW, float* __restrict__ Hgc) {
  long idx = (long)blockIdx.x * blockDim.x + threadIdx.x;
  if (idx >= (long)NNZ_E * TT) return;
  int e = (int)(idx / TT);
  int t = (int)(idx % TT);
  int r = rows[e], c = cols[e];
  float v = vals[e];
  const float4* src = (const float4*)(HW + ((long)c * TT + t) * DM);
  float* dst = Hgc + ((long)r * TT + t) * DM;
#pragma unroll
  for (int j = 0; j < 16; ++j) {
    float4 x = src[j];
    atomicAdd(dst + j * 4 + 0, v * x.x);
    atomicAdd(dst + j * 4 + 1, v * x.y);
    atomicAdd(dst + j * 4 + 2, v * x.z);
    atomicAdd(dst + j * 4 + 3, v * x.w);
  }
}

// ---------------------------------------------------------------------------
// Kernel 4: per node (one wave): x = relu(Hgc+gc_b) in LDS (12x64 f16);
// causal width-3 conv as two 16x192x64 WMMA GEMMs with A fragments gathered
// DIRECTLY from the compact x buffer (each 8-half run of an A fragment is 8
// consecutive channels of one time-row ts=m+k-2, never crossing a tap
// boundary) -> gate -> residual -> LayerNorm (shfl_xor) -> H in place.
// 128 threads = 4 waves; one node per wave.
// ---------------------------------------------------------------------------
__global__ void k_conv_ln(const float* __restrict__ Hgc, const float* __restrict__ Hres,
                          const float* __restrict__ cfw, const float* __restrict__ cfb,
                          const float* __restrict__ cgw, const float* __restrict__ cgb,
                          const float* __restrict__ gcb,
                          const float* __restrict__ lng, const float* __restrict__ lnb,
                          float* __restrict__ Hout) {
  __shared__ __attribute__((aligned(32))) _Float16 sBf[192 * 64];   // kc=6, nt=4
  __shared__ __attribute__((aligned(32))) _Float16 sBg[192 * 64];
  __shared__ __attribute__((aligned(32))) _Float16 sX[4][TT * DM];  // 12x64 f16/wave
  __shared__ float sCfb[64], sCgb[64], sGcb[64], sLg[64], sLb[64];

  int tid = threadIdx.x;
  // B row index K = k*64+din, col = dout ; weights (64,64,3) row-major.
  for (int i = tid; i < 24 * 512; i += 128) {
    int tile = i >> 9, rem = i & 511;
    int lane2 = rem >> 4, h = rem & 15;
    int kc = tile >> 2, nt = tile & 3;
    int K = kc * 32 + ((h >> 3) << 4) + ((lane2 >> 4) << 3) + (h & 7);
    int dout = nt * 16 + (lane2 & 15);
    int k = K >> 6, din = K & 63;
    long widx = ((long)dout * 64 + din) * 3 + k;
    sBf[i] = (_Float16)cfw[widx];
    sBg[i] = (_Float16)cgw[widx];
  }
  for (int i = tid; i < 64; i += 128) {
    sCfb[i] = cfb[i]; sCgb[i] = cgb[i]; sGcb[i] = gcb[i];
    sLg[i] = lng[i];  sLb[i] = lnb[i];
  }
  __syncthreads();

  int wave = tid >> 5, lane = tid & 31;
  int node = blockIdx.x * 4 + wave;
  if (node >= NNODES) return;

  // x = relu(Hgc + gc_b) -> f16 LDS (12x64): 6 float4 loads + 6 b64 stores/lane
  _Float16* X = sX[wave];
  const float4* src4 = (const float4*)(Hgc + (long)node * TT * DM);
  for (int q = lane; q < (TT * DM) / 4; q += 32) {
    int col = (q * 4) & 63;
    float4 xx = src4[q];
    float f0 = xx.x + sGcb[col + 0]; f0 = f0 > 0.f ? f0 : 0.f;
    float f1 = xx.y + sGcb[col + 1]; f1 = f1 > 0.f ? f1 : 0.f;
    float f2 = xx.z + sGcb[col + 2]; f2 = f2 > 0.f ? f2 : 0.f;
    float f3 = xx.w + sGcb[col + 3]; f3 = f3 > 0.f ? f3 : 0.f;
    v4h hv;
    hv[0] = (_Float16)f0; hv[1] = (_Float16)f1;
    hv[2] = (_Float16)f2; hv[3] = (_Float16)f3;
    *(v4h*)(X + q * 4) = hv;
  }

  int m    = lane & 15;
  int ksel = (lane >> 4) * 8;

  // A fragments for K chunks 0..5, gathered from X with causal shift.
  v16h afr[6];
#pragma unroll
  for (int kc = 0; kc < 6; ++kc) {
    v16h a;
#pragma unroll
    for (int r = 0; r < 2; ++r) {
      int kd  = kc * 32 + r * 16 + ksel;   // composite K index = k*64 + din
      int k   = kd >> 6;
      int din = kd & 63;
      int ts  = m + k - 2;
      v8h v;
      if (ts >= 0 && ts < TT) {
        v = *(const v8h*)(X + ts * DM + din);
      } else {
#pragma unroll
        for (int j = 0; j < 8; ++j) v[j] = (_Float16)0.f;
      }
#pragma unroll
      for (int j = 0; j < 8; ++j) a[r * 8 + j] = v[j];
    }
    afr[kc] = a;
  }

  int n = lane & 15, mh = (lane >> 4) * 8;
  v8f yt[4];
#pragma unroll
  for (int nt = 0; nt < 4; ++nt) {
    v8f f = {}, g = {};
#pragma unroll
    for (int kc = 0; kc < 6; ++kc) {
      f = wmma32(afr[kc], load_b_sw(sBf, 4, kc, nt), f);
      g = wmma32(afr[kc], load_b_sw(sBg, 4, kc, nt), g);
    }
    float bf = sCfb[nt * 16 + n], bg = sCgb[nt * 16 + n];
#pragma unroll
    for (int i = 0; i < 8; ++i) {
      float ff = tanhf(f[i] + bf);
      float gg = g[i] + bg;
      gg = 1.f / (1.f + __expf(-gg));
      int t = mh + i;
      float res = (t < TT) ? Hres[((long)node * TT + t) * DM + nt * 16 + n] : 0.f;
      yt[nt][i] = ff * gg + res;
    }
  }

  // LayerNorm over 64 channels of row t = mh+i (one 16-lane half-group x 4
  // column tiles) via shfl_xor reductions.
#pragma unroll
  for (int i = 0; i < 8; ++i) {
    float s1 = yt[0][i] + yt[1][i] + yt[2][i] + yt[3][i];
    float s2 = yt[0][i] * yt[0][i] + yt[1][i] * yt[1][i] +
               yt[2][i] * yt[2][i] + yt[3][i] * yt[3][i];
#pragma unroll
    for (int mm = 1; mm < 16; mm <<= 1) {
      s1 += __shfl_xor(s1, mm, 32);
      s2 += __shfl_xor(s2, mm, 32);
    }
    float mu   = s1 * (1.f / 64.f);
    float var  = s2 * (1.f / 64.f) - mu * mu;
    float rstd = rsqrtf(var + 1e-5f);
    int t = mh + i;
    if (t < TT) {
#pragma unroll
      for (int nt = 0; nt < 4; ++nt) {
        int col = nt * 16 + n;
        Hout[((long)node * TT + t) * DM + col] =
            (yt[nt][i] - mu) * rstd * sLg[col] + sLb[col];
      }
    }
  }
}

// ---------------------------------------------------------------------------
// Kernel 5: head: relu(H[:, -1, :] @ W1 + b1) @ W2 + b2 -> out (N,)
// ---------------------------------------------------------------------------
__global__ void k_head(const float* __restrict__ Hfin,
                       const float* __restrict__ w1, const float* __restrict__ b1,
                       const float* __restrict__ w2, const float* __restrict__ b2,
                       float* __restrict__ out) {
  __shared__ __attribute__((aligned(32))) _Float16 sW1[64 * 32];   // kc=2, nt=2
  __shared__ float sB1[32], sW2f[32];
  __shared__ float sB2;
  __shared__ __attribute__((aligned(32))) _Float16 sHid[4][16 * 32];

  int tid = threadIdx.x;
  fill_b_sw(sW1, w1, 64, 32, 64, 128);
  for (int i = tid; i < 32; i += 128) { sB1[i] = b1[i]; sW2f[i] = w2[i]; }
  if (tid == 0) sB2 = b2[0];
  __syncthreads();

  int wave = tid >> 5, lane = tid & 31;
  int tile = blockIdx.x * 4 + wave;
  if (tile * 16 >= NNODES) return;
  int row0 = tile * 16;

  // A straight from global: rows stride TT*64 floats, last timestep.
  const float* rb = Hfin + ((long)row0 * TT + (TT - 1)) * DM;
  v16h a0 = load_a_g32(rb, TT * DM, 0);
  v16h a1 = load_a_g32(rb, TT * DM, 32);
  _Float16* Hd = sHid[wave];
#pragma unroll
  for (int nt = 0; nt < 2; ++nt) {
    v8f acc = {};
    acc = wmma32(a0, load_b_sw(sW1, 2, 0, nt), acc);
    acc = wmma32(a1, load_b_sw(sW1, 2, 1, nt), acc);
    acc = addbias(acc, sB1, nt * 16);
    store_d16(Hd, 32, nt * 16, acc, true);
  }
  if (lane < 16) {
    float acc = sB2;
    const v8h* hp = (const v8h*)(Hd + lane * 32);
#pragma unroll
    for (int cc = 0; cc < 4; ++cc) {
      v8h hv = hp[cc];
#pragma unroll
      for (int j = 0; j < 8; ++j) acc += (float)hv[j] * sW2f[cc * 8 + j];
    }
    out[row0 + lane] = acc;
  }
}

// ---------------------------------------------------------------------------
extern "C" void kernel_launch(void* const* d_in, const int* in_sizes, int n_in,
                              void* d_out, int out_size, void* d_ws, size_t ws_size,
                              hipStream_t stream) {
  (void)in_sizes; (void)n_in; (void)out_size; (void)ws_size;

  const float* Xe    = (const float*)d_in[0];
  const float* Xw    = (const float*)d_in[1];
  const int*   arows = (const int*)  d_in[2];
  const int*   acols = (const int*)  d_in[3];
  const float* avals = (const float*)d_in[4];
  const float* ee_w1 = (const float*)d_in[5];
  const float* ee_b1 = (const float*)d_in[6];
  const float* ee_w2 = (const float*)d_in[7];
  const float* ee_b2 = (const float*)d_in[8];
  const float* we_w1 = (const float*)d_in[9];
  const float* we_b1 = (const float*)d_in[10];
  const float* we_w2 = (const float*)d_in[11];
  const float* we_b2 = (const float*)d_in[12];
  const float* comb_w = (const float*)d_in[13];
  const float* comb_b = (const float*)d_in[14];
  const float* gc_W  = (const float*)d_in[15];
  const float* gc_b  = (const float*)d_in[16];
  const float* cf_w  = (const float*)d_in[17];
  const float* cf_b  = (const float*)d_in[18];
  const float* cg_w  = (const float*)d_in[19];
  const float* cg_b  = (const float*)d_in[20];
  const float* ln_g  = (const float*)d_in[21];
  const float* ln_b  = (const float*)d_in[22];
  const float* ph_w1 = (const float*)d_in[23];
  const float* ph_b1 = (const float*)d_in[24];
  const float* ph_w2 = (const float*)d_in[25];
  const float* ph_b2 = (const float*)d_in[26];

  float* H   = (float*)d_ws;                 // 240000*64 f32 = 61.44 MB
  float* HW  = H  + (long)NROWS * DM;        // 61.44 MB
  float* Hgc = HW + (long)NROWS * DM;        // 61.44 MB

  // 1) fused embeddings -> H
  k_embed<<<NROWS / 16 / 4, 128, 0, stream>>>(Xe, Xw, ee_w1, ee_b1, ee_w2, ee_b2,
                                              we_w1, we_b1, we_w2, we_b2,
                                              comb_w, comb_b, H);

  for (int blk = 0; blk < 2; ++blk) {
    // 2) HW = H @ gc_W[blk]
    k_gemm64<<<1875, 128, 0, stream>>>(H, gc_W + (long)blk * DM * DM, HW);
    // 3) Hgc = segment_sum(adj_vals * HW[cols]) via L2 atomics
    hipMemsetAsync(Hgc, 0, (size_t)NROWS * DM * sizeof(float), stream);
    k_scatter<<<(int)(((long)NNZ_E * TT + 255) / 256), 256, 0, stream>>>(
        arows, acols, avals, HW, Hgc);
    // 4) conv + gate + residual + LN  (H updated in place)
    k_conv_ln<<<NNODES / 4, 128, 0, stream>>>(
        Hgc, H,
        cf_w + (long)blk * DM * DM * 3, cf_b + (long)blk * DM,
        cg_w + (long)blk * DM * DM * 3, cg_b + (long)blk * DM,
        gc_b + (long)blk * DM,
        ln_g + (long)blk * DM, ln_b + (long)blk * DM, H);
  }

  // 5) head -> out (N,)
  k_head<<<(NNODES / 16 + 3) / 4, 128, 0, stream>>>(H, ph_w1, ph_b1, ph_w2, ph_b2,
                                                    (float*)d_out);
}
